// CrossDepthAttentionResidual_72619307041485
// MI455X (gfx1250) — matrix operand order repaired and checked by reading, not compiled, not run on hardware
//
#include <hip/hip_runtime.h>
#include <hip/hip_bf16.h>
#include <math.h>

typedef __attribute__((ext_vector_type(16))) __bf16 v16bf;
typedef __attribute__((ext_vector_type(8)))  __bf16 v8bf;
typedef __attribute__((ext_vector_type(8)))  float  v8f;
typedef __attribute__((ext_vector_type(4)))  unsigned int v4u;
typedef __attribute__((ext_vector_type(8)))  int v8i;
typedef __attribute__((ext_vector_type(4)))  int v4i;

namespace {
constexpr int kL    = 12;     // layers
constexpr int kTok  = 4096;   // B*S tokens
constexpr int kD    = 1024;   // hidden dim
constexpr int kDK   = 256;    // key dim
constexpr int kT    = 16;     // tokens per workgroup (WMMA M)
constexpr int kPad  = kD + 8; // padded bf16 LDS row stride (bank-conflict-free b128)
constexpr float kEps = 1e-5f;

constexpr size_t kSmemBytes =
    (size_t)kT * kD * 4 * 2    // sRaw0 + sRaw1 (f32 raw tiles, contiguous for TDM)
  + (size_t)kT * kD * 4        // sAcc   (f32 online-softmax accumulator)
  + (size_t)kT * kDK * 4       // sQ     (f32 query tile)
  + (size_t)kD * 4 * 2         // sLnw + sLnb
  + (size_t)kT * kPad * 2 * 2  // sNorm0 + sNorm1 (bf16, padded)
  + (size_t)kT * 4 * 7;        // sLogit0/1, sM, sSum, sAlpha, sW0, sW1
} // namespace

// ---------------- weight fp32 -> bf16 conversion (runs once per launch) ----
__global__ void cvt_weights_kernel(const float* __restrict__ Wq,
                                   const float* __restrict__ Wk,
                                   __bf16* __restrict__ q_bf,
                                   __bf16* __restrict__ k_bf) {
  int i = blockIdx.x * blockDim.x + threadIdx.x;   // [0, kDK*kD)
  q_bf[i] = (__bf16)Wq[i];
  k_bf[i] = (__bf16)Wk[i];
}

// ---- generic -> LDS address-space offset ----------------------------------
__device__ inline unsigned lds_offset_of(const void* p) {
  typedef __attribute__((address_space(3))) const void lds_cv;
  return (unsigned)(unsigned long long)(lds_cv*)p;
}

// ---- Tensor Data Mover: load `nlayers` 16x1024 f32 tiles (stride Tok*D) ---
// D# per ISA ch.8: group0 = {count=1, lds_addr, global_addr, type=2};
// group1 = {data_size=4B, tensor_dim0=16384, tensor_dim1=nlayers,
//           tile_dim0=16384, tile_dim1=nlayers, dim0_stride=Tok*D}.
// LDS fill is packed-sequential, so the tiles land back-to-back.
__device__ inline void tdm_load_tiles(const float* __restrict__ gsrc,
                                      unsigned lds_off, int nlayers) {
  unsigned long long ga = (unsigned long long)(uintptr_t)gsrc;
  v4u g0 = { 1u,                                          // count=1 (valid user D#)
             lds_off,                                     // lds_addr
             (unsigned)(ga & 0xFFFFFFFFull),              // global_addr[31:0]
             (unsigned)((ga >> 32) & 0x01FFFFFFull)       // global_addr[56:32]
               | 0x80000000u };                           // type=2 ("image")
  v8i g1 = { (int)0x00020000u,                            // data_size=4B; no flags
             (int)(16384u << 16),                         // tensor_dim0[15:0]=16384
             (int)((unsigned)nlayers << 16),              // dim0[31:16]=0 | tensor_dim1=nlayers
             (int)(16384u << 16),                         // dim1[31:16]=0 | tile_dim0=16384
             nlayers,                                     // tile_dim1 | tile_dim2=0
             (int)(kTok * kD),                            // tensor_dim0_stride[31:0]
             0, 0 };                                      // stride hi / dim1_stride
  v4i gz4 = {0, 0, 0, 0};
  v8i gz8 = {0, 0, 0, 0, 0, 0, 0, 0};
  // 6-arg form (clang-23 / therock-10.0 headers): groups 2,3 + extra group, cpol
  __builtin_amdgcn_tensor_load_to_lds(g0, g1, gz4, gz4, gz8, 0);
}

// ---- A fragment: 16x32 bf16 tile from padded LDS (contiguous k = 16h + s order) ----
__device__ inline v16bf load_a_frag(const __bf16* sNorm, int kb, int lane) {
  const int m    = lane & 15;
  const int koff = (lane & 16) ? 16 : 0;           // lanes 16-31: K=16..31
  const __bf16* p = sNorm + m * kPad + kb * 32 + koff;
  v16bf a;
  *reinterpret_cast<v8bf*>(&a)       = *reinterpret_cast<const v8bf*>(p);       // VGPR0-3
  *(reinterpret_cast<v8bf*>(&a) + 1) = *reinterpret_cast<const v8bf*>(p + 8);   // VGPR4-7
  return a;
}

// ---- B fragment: 32x16 bf16 = W^T tile; W row-major [kDK][kD] bf16 --------
__device__ inline v16bf load_b_frag(const __bf16* __restrict__ W,
                                    int ntile, int kb, int lane) {
  const int n    = ntile * 16 + (lane & 15);
  const int koff = (lane & 16) ? 16 : 0;           // lanes 16-31: K=16..31
  const __bf16* p = W + (size_t)n * kD + kb * 32 + koff;
  return *reinterpret_cast<const v16bf*>(p);       // 32B contiguous -> 2x b128
}

// ---- LayerNorm: wave `wave` normalizes rows 2w, 2w+1 of one tile ----------
__device__ inline void layernorm_rows(const float* sRaw, __bf16* sNorm,
                                      const float* sLnw, const float* sLnb,
                                      int wave, int lane) {
#pragma unroll
  for (int rr = 0; rr < 2; ++rr) {
    const int r = wave * 2 + rr;
    const float* row = sRaw + r * kD;
    float s1 = 0.f, s2 = 0.f;
#pragma unroll
    for (int i = 0; i < kD / 32; ++i) {
      float x = row[lane + 32 * i];
      s1 += x; s2 += x * x;
    }
    for (int m = 16; m >= 1; m >>= 1) {
      s1 += __shfl_xor(s1, m, 32);
      s2 += __shfl_xor(s2, m, 32);
    }
    const float mu  = s1 * (1.0f / kD);
    const float var = s2 * (1.0f / kD) - mu * mu;
    const float rs  = rsqrtf(var + kEps);
    __bf16* nrow = sNorm + r * kPad;
#pragma unroll
    for (int i = 0; i < kD / 32; ++i) {
      const int k = lane + 32 * i;
      nrow[k] = (__bf16)((row[k] - mu) * rs * sLnw[k] + sLnb[k]);
    }
  }
}

// ---- single-tile GEMM (query path): two N-tiles per wave ------------------
__device__ inline void gemm_one(const __bf16* sNorm, const __bf16* __restrict__ W,
                                int nt0, int nt1, int lane, v8f& c0, v8f& c1) {
  c0 = (v8f){0.f,0.f,0.f,0.f,0.f,0.f,0.f,0.f};
  c1 = (v8f){0.f,0.f,0.f,0.f,0.f,0.f,0.f,0.f};
#pragma unroll 4
  for (int kb = 0; kb < kD / 32; ++kb) {
    v16bf a  = load_a_frag(sNorm, kb, lane);
    v16bf b0 = load_b_frag(W, nt0, kb, lane);
    v16bf b1 = load_b_frag(W, nt1, kb, lane);
    c0 = __builtin_amdgcn_wmma_f32_16x16x32_bf16(false, a, false, b0, (short)0, c0, false, false);
    c1 = __builtin_amdgcn_wmma_f32_16x16x32_bf16(false, a, false, b1, (short)0, c1, false, false);
  }
}

// ---- paired GEMM: each B fragment feeds both layers' A tiles (2x reuse) ---
__device__ inline void gemm_pair(const __bf16* sN0, const __bf16* sN1,
                                 const __bf16* __restrict__ W,
                                 int nt0, int nt1, int lane,
                                 v8f& c00, v8f& c01, v8f& c10, v8f& c11) {
  c00 = (v8f){0.f,0.f,0.f,0.f,0.f,0.f,0.f,0.f};
  c01 = c00; c10 = c00; c11 = c00;
#pragma unroll 2
  for (int kb = 0; kb < kD / 32; ++kb) {
    v16bf a0 = load_a_frag(sN0, kb, lane);
    v16bf a1 = load_a_frag(sN1, kb, lane);
    v16bf b0 = load_b_frag(W, nt0, kb, lane);
    v16bf b1 = load_b_frag(W, nt1, kb, lane);
    c00 = __builtin_amdgcn_wmma_f32_16x16x32_bf16(false, a0, false, b0, (short)0, c00, false, false);
    c01 = __builtin_amdgcn_wmma_f32_16x16x32_bf16(false, a0, false, b1, (short)0, c01, false, false);
    c10 = __builtin_amdgcn_wmma_f32_16x16x32_bf16(false, a1, false, b0, (short)0, c10, false, false);
    c11 = __builtin_amdgcn_wmma_f32_16x16x32_bf16(false, a1, false, b1, (short)0, c11, false, false);
  }
}

// ---------------------------------------------------------------------------
__global__ __launch_bounds__(256, 1)
void cross_depth_attn_kernel(const float* __restrict__ states,   // [L, tok, D]
                             const __bf16* __restrict__ Wq_bf,   // [DK, D]
                             const __bf16* __restrict__ Wk_bf,   // [DK, D]
                             const float* __restrict__ lnw,
                             const float* __restrict__ lnb,
                             const float* __restrict__ gate_in,  // scalar
                             float* __restrict__ out) {          // [tok, D]
  extern __shared__ char smem[];
  float*  sRaw0  = reinterpret_cast<float*>(smem);     // pair buffers, contiguous
  float*  sRaw1  = sRaw0 + kT * kD;
  float*  sAcc   = sRaw1 + kT * kD;
  float*  sQ     = sAcc + kT * kD;
  float*  sLnw   = sQ + kT * kDK;
  float*  sLnb   = sLnw + kD;
  __bf16* sNorm0 = reinterpret_cast<__bf16*>(sLnb + kD);
  __bf16* sNorm1 = sNorm0 + kT * kPad;
  float*  sLogit0 = reinterpret_cast<float*>(sNorm1 + kT * kPad);
  float*  sLogit1 = sLogit0 + kT;
  float*  sM     = sLogit1 + kT;
  float*  sSum   = sM + kT;
  float*  sAlpha = sSum + kT;
  float*  sW0    = sAlpha + kT;
  float*  sW1    = sW0 + kT;

  const int tid   = threadIdx.x;
  const int lane  = tid & 31;
  const int wave  = tid >> 5;
  const int nt0   = wave * 2;
  const int nt1   = wave * 2 + 1;
  const int tile0 = blockIdx.x * kT;   // first token of this tile
  const int roff  = (lane & 16) ? 8 : 0;
  const int nl    = lane & 15;
  const unsigned raw_lds = lds_offset_of(sRaw0);

  // init: LN params, accumulator, softmax state; kick TDM for layer 11 tile
  if (wave == 0) {
    tdm_load_tiles(states + ((size_t)(kL - 1) * kTok + tile0) * kD, raw_lds, 1);
  }
  for (int i = tid; i < kD; i += 256) { sLnw[i] = lnw[i]; sLnb[i] = lnb[i]; }
  float4* acc4 = reinterpret_cast<float4*>(sAcc);
  for (int i = tid; i < kT * kD / 4; i += 256) acc4[i] = make_float4(0.f, 0.f, 0.f, 0.f);
  if (tid < kT) { sM[tid] = -INFINITY; sSum[tid] = 0.f; }
  if (wave == 0) __builtin_amdgcn_s_wait_tensorcnt(0);
  __syncthreads();

  // ---- phase A: layer L-1 -> LayerNorm -> query tile via WMMA -------------
  layernorm_rows(sRaw0, sNorm0, sLnw, sLnb, wave, lane);
  __syncthreads();
  {
    v8f c0, c1;
    gemm_one(sNorm0, Wq_bf, nt0, nt1, lane, c0, c1);
#pragma unroll
    for (int g = 0; g < 8; ++g) {            // C layout: VGPR g -> row g (+8 hi lanes)
      sQ[(g + roff) * kDK + nt0 * 16 + nl] = c0[g];
      sQ[(g + roff) * kDK + nt1 * 16 + nl] = c1[g];
    }
  }

  const float scale = 0.0625f;               // 1/sqrt(DK)

  // ---- stream layer pairs (0,1)..(10,11); layer 11 ends resident in sRaw1 -
  for (int lp = 0; lp < kL / 2; ++lp) {
    const int l = lp * 2;
    __syncthreads();                          // raw buffers free, sQ visible
    if (wave == 0) {                          // one TDM moves both 64KB tiles
      tdm_load_tiles(states + ((size_t)l * kTok + tile0) * kD, raw_lds, 2);
    }
    if (tid < kT) { sLogit0[tid] = 0.f; sLogit1[tid] = 0.f; }
    if (l + 2 < kL) {                         // pull next pair toward L2
      const char* nxt = reinterpret_cast<const char*>(
          states + ((size_t)(l + 2) * kTok + tile0) * kD);
      __builtin_prefetch(nxt + tid * 512, 0, 1);
      __builtin_prefetch(nxt + tid * 512 + 256, 0, 1);
    }
    if (wave == 0) __builtin_amdgcn_s_wait_tensorcnt(0);
    __syncthreads();

    layernorm_rows(sRaw0, sNorm0, sLnw, sLnb, wave, lane);
    layernorm_rows(sRaw1, sNorm1, sLnw, sLnb, wave, lane);
    __syncthreads();

    // paired keys GEMM (WMMA) + per-token logits for both layers
    v8f c00, c01, c10, c11;
    gemm_pair(sNorm0, sNorm1, Wk_bf, nt0, nt1, lane, c00, c01, c10, c11);
#pragma unroll
    for (int g = 0; g < 8; ++g) {
      const float q0 = sQ[(g + roff) * kDK + nt0 * 16 + nl];
      const float q1 = sQ[(g + roff) * kDK + nt1 * 16 + nl];
      float va = c00[g] * q0 + c01[g] * q1;
      float vb = c10[g] * q0 + c11[g] * q1;
      for (int m = 8; m >= 1; m >>= 1) {      // 16-lane groups share a token row
        va += __shfl_xor(va, m, 32);
        vb += __shfl_xor(vb, m, 32);
      }
      if (nl == 0) {
        atomicAdd(&sLogit0[g + roff], va);    // ds_add_f32
        atomicAdd(&sLogit1[g + roff], vb);
      }
    }
    __syncthreads();

    // combined online-softmax update for the pair (one thread per token row)
    if (tid < kT) {
      const float l0 = sLogit0[tid] * scale;
      const float l1 = sLogit1[tid] * scale;
      const float mOld = sM[tid];
      const float mNew = fmaxf(mOld, fmaxf(l0, l1));
      const float alpha = __expf(mOld - mNew);
      const float w0 = __expf(l0 - mNew);
      const float w1 = __expf(l1 - mNew);
      sSum[tid] = sSum[tid] * alpha + w0 + w1;
      sM[tid]   = mNew;
      sAlpha[tid] = alpha;
      sW0[tid] = w0;
      sW1[tid] = w1;
    }
    __syncthreads();

    // rescale accumulator and fold in both layers' raw states
    const float4* s04 = reinterpret_cast<const float4*>(sRaw0);
    const float4* s14 = reinterpret_cast<const float4*>(sRaw1);
    for (int i = tid; i < kT * kD / 4; i += 256) {
      const int r = i >> 8;                  // 256 float4 per row
      const float a = sAlpha[r], w0 = sW0[r], w1 = sW1[r];
      float4 s0 = s04[i], s1 = s14[i], u = acc4[i];
      u.x = u.x * a + w0 * s0.x + w1 * s1.x;
      u.y = u.y * a + w0 * s0.y + w1 * s1.y;
      u.z = u.z * a + w0 * s0.z + w1 * s1.z;
      u.w = u.w * a + w0 * s0.w + w1 * s1.w;
      acc4[i] = u;
    }
  }

  __syncthreads();
  // ---- gated residual: out = g*states[11] + (1-g)*acc/sum -----------------
  const float gv = 1.f / (1.f + __expf(-gate_in[0]));
  const float4* st4 = reinterpret_cast<const float4*>(sRaw1);     // layer 11
  float4* o4 = reinterpret_cast<float4*>(out + (size_t)tile0 * kD);
  for (int i = tid; i < kT * kD / 4; i += 256) {
    const int r = i >> 8;
    const float inv = 1.f / sSum[r];
    float4 s = st4[i], a = acc4[i], o;
    o.x = gv * s.x + (1.f - gv) * a.x * inv;
    o.y = gv * s.y + (1.f - gv) * a.y * inv;
    o.z = gv * s.z + (1.f - gv) * a.z * inv;
    o.w = gv * s.w + (1.f - gv) * a.w * inv;
    o4[i] = o;
  }
}

// ---------------------------------------------------------------------------
extern "C" void kernel_launch(void* const* d_in, const int* in_sizes, int n_in,
                              void* d_out, int out_size, void* d_ws, size_t ws_size,
                              hipStream_t stream) {
  (void)in_sizes; (void)n_in; (void)out_size; (void)ws_size;
  const float* states = (const float*)d_in[0];   // (L,B,S,D) f32
  const float* Wq     = (const float*)d_in[1];   // (DK,D) f32
  const float* Wk     = (const float*)d_in[2];   // (DK,D) f32
  const float* lnw    = (const float*)d_in[3];   // (D,)
  const float* lnb    = (const float*)d_in[4];   // (D,)
  const float* gate   = (const float*)d_in[5];   // scalar
  float* out          = (float*)d_out;           // (B,S,D) f32

  __bf16* q_bf = (__bf16*)d_ws;                  // 512 KB
  __bf16* k_bf = q_bf + (size_t)kDK * kD;        // 512 KB

  cvt_weights_kernel<<<(kDK * kD) / 256, 256, 0, stream>>>(Wq, Wk, q_bf, k_bf);

  cross_depth_attn_kernel<<<kTok / kT, 256, kSmemBytes, stream>>>(
      states, q_bf, k_bf, lnw, lnb, gate, out);
}
